// Model_31026843746798
// MI455X (gfx1250) — matrix operand order repaired and checked by reading, not compiled, not run on hardware
//
#include <hip/hip_runtime.h>
#include <hip/hip_bf16.h>
#include <math.h>

// ---------------- model constants ----------------
#define Bb      4
#define Ll      1024
#define Dd      256
#define NLAYERS 4
#define NEXP    4
#define DSTATE  16
#define DCONV   4
#define DINN    512
#define HH      1024
#define DTRR    16
#define RR      (Bb * Ll)      // 4096 tokens
#define XZD     (2 * DINN)     // 1024

typedef _Float16 half_t;
typedef __attribute__((ext_vector_type(16))) _Float16 v16h;
typedef __attribute__((ext_vector_type(8)))  _Float16 v8h;
typedef __attribute__((ext_vector_type(8)))  float    v8f;

// ---------------- WMMA fragment loaders (CDNA5 wave32 layouts) ----------------
// A fragment: 16x32 f16, lane = m (0..15) + half-select s. Lane holds
// K = 8s + {0..7} and 16 + 8s + {0..7}  -> two contiguous 16B runs.
__device__ __forceinline__ v16h load_a(const half_t* __restrict__ p0, int lda, int lane) {
    int m = lane & 15, s = lane >> 4;
    const half_t* p = p0 + m * lda + 8 * s;
    v8h lo = *(const v8h*)(p);
    v8h hi = *(const v8h*)(p + 16);
    v16h r;
#pragma unroll
    for (int i = 0; i < 8; ++i) { r[i] = lo[i]; r[i + 8] = hi[i]; }
    return r;
}

// B fragment: 32x16 f16 = W^T tile from row-major W[N][K]. Lane = n, holds
// K = 8s + {0..7} and 16 + 8s + {0..7} -> two contiguous 16B runs of row n of W.
__device__ __forceinline__ v16h load_b(const half_t* __restrict__ p0, int ldw, int lane) {
    int n = lane & 15, s = lane >> 4;
    const half_t* p = p0 + n * ldw + 8 * s;
    v8h lo = *(const v8h*)(p);
    v8h hi = *(const v8h*)(p + 16);
    v16h r;
#pragma unroll
    for (int i = 0; i < 8; ++i) { r[i] = lo[i]; r[i + 8] = hi[i]; }
    return r;
}

#define WMMA_F16(a, b, c) \
    __builtin_amdgcn_wmma_f32_16x16x32_f16(false, (a), false, (b), (short)0, (c), false, false)

// ---------------- generic GEMM: C(f32, MxN) = A(f16, MxK) * W(f16, NxK)^T ----------------
// 2x(NT) register blocking; per k-step: issue all loads (one clause), then all
// 2*NT WMMAs back-to-back (single loadcnt wait, uninterrupted XDL burst).
template <int NT>
__global__ void gemm_wmma(float* __restrict__ C, const half_t* __restrict__ A,
                          const half_t* __restrict__ W, int M, int N, int K) {
    int lane = threadIdx.x & 31;
    int wv   = blockIdx.x * 8 + (threadIdx.x >> 5);
    int ngroups = N / (16 * NT);
    int mt = wv / ngroups;
    int ng = wv - mt * ngroups;
    if (mt >= M / 32) return;
    int m0 = mt * 32, n0 = ng * 16 * NT;

    v8f acc0[NT] = {}, acc1[NT] = {};
    const half_t* Arow0 = A + (size_t)m0 * K;
    const half_t* Arow1 = A + (size_t)(m0 + 16) * K;
    for (int k0 = 0; k0 < K; k0 += 32) {
        __builtin_prefetch(Arow0 + ((lane & 15) * K) + k0 + 64, 0, 1);
        v16h a0 = load_a(Arow0 + k0, K, lane);
        v16h a1 = load_a(Arow1 + k0, K, lane);
        v16h bf[NT];
#pragma unroll
        for (int t = 0; t < NT; ++t)
            bf[t] = load_b(W + (size_t)(n0 + 16 * t) * K + k0, K, lane);
#pragma unroll
        for (int t = 0; t < NT; ++t) {
            acc0[t] = WMMA_F16(a0, bf[t], acc0[t]);
            acc1[t] = WMMA_F16(a1, bf[t], acc1[t]);
        }
    }
    int n = lane & 15, s = lane >> 4;
#pragma unroll
    for (int t = 0; t < NT; ++t) {
        float* Cp0 = C + (size_t)m0 * N + n0 + 16 * t + n;
        float* Cp1 = C + (size_t)(m0 + 16) * N + n0 + 16 * t + n;
#pragma unroll
        for (int r = 0; r < 8; ++r) {
            Cp0[(size_t)(r + 8 * s) * N] = acc0[t][r];
            Cp1[(size_t)(r + 8 * s) * N] = acc1[t][r];
        }
    }
}

// ---------------- expert-up GEMM with bias + exact GELU, f16 out ----------------
template <int NT>
__global__ void gemm_wmma_gelu(half_t* __restrict__ Cout, const half_t* __restrict__ A,
                               const half_t* __restrict__ W, const float* __restrict__ bias,
                               int M, int N, int K) {
    int lane = threadIdx.x & 31;
    int wv   = blockIdx.x * 8 + (threadIdx.x >> 5);
    int ngroups = N / (16 * NT);
    int mt = wv / ngroups;
    int ng = wv - mt * ngroups;
    if (mt >= M / 32) return;
    int m0 = mt * 32, n0 = ng * 16 * NT;

    v8f acc0[NT] = {}, acc1[NT] = {};
    const half_t* Arow0 = A + (size_t)m0 * K;
    const half_t* Arow1 = A + (size_t)(m0 + 16) * K;
    for (int k0 = 0; k0 < K; k0 += 32) {
        __builtin_prefetch(Arow0 + ((lane & 15) * K) + k0 + 64, 0, 1);
        v16h a0 = load_a(Arow0 + k0, K, lane);
        v16h a1 = load_a(Arow1 + k0, K, lane);
        v16h bf[NT];
#pragma unroll
        for (int t = 0; t < NT; ++t)
            bf[t] = load_b(W + (size_t)(n0 + 16 * t) * K + k0, K, lane);
#pragma unroll
        for (int t = 0; t < NT; ++t) {
            acc0[t] = WMMA_F16(a0, bf[t], acc0[t]);
            acc1[t] = WMMA_F16(a1, bf[t], acc1[t]);
        }
    }
    int n = lane & 15, s = lane >> 4;
#pragma unroll
    for (int t = 0; t < NT; ++t) {
        float bb = bias[n0 + 16 * t + n];
#pragma unroll
        for (int r = 0; r < 8; ++r) {
            float v0 = acc0[t][r] + bb;
            float v1 = acc1[t][r] + bb;
            v0 = 0.5f * v0 * (1.0f + erff(v0 * 0.70710678118654752f));  // exact GELU
            v1 = 0.5f * v1 * (1.0f + erff(v1 * 0.70710678118654752f));
            Cout[(size_t)(m0 + r + 8 * s) * N + n0 + 16 * t + n]      = (half_t)v0;
            Cout[(size_t)(m0 + 16 + r + 8 * s) * N + n0 + 16 * t + n] = (half_t)v1;
        }
    }
}

// ---------------- MoE down-proj: residual += sum_e gate_e * (H_e * W2_e^T + b2_e) ----------------
__global__ void mk_moe2(float* __restrict__ residual, const half_t* __restrict__ Hbuf,
                        const half_t* __restrict__ W2, const float* __restrict__ b2,
                        const float* __restrict__ gate) {
    const int K = HH, N = Dd;
    int lane = threadIdx.x & 31;
    int wv   = blockIdx.x * 8 + (threadIdx.x >> 5);
    int mt = wv >> 2, ng = wv & 3;           // ngroups = 256/64 = 4
    if (mt >= RR / 32) return;
    int m0 = mt * 32, n0 = ng * 64;
    int n = lane & 15, s = lane >> 4;

    v8f tot0[4] = {}, tot1[4] = {};
    for (int e = 0; e < NEXP; ++e) {
        v8f acc0[4] = {}, acc1[4] = {};
        const half_t* A0 = Hbuf + (size_t)e * RR * K + (size_t)m0 * K;
        const half_t* A1 = A0 + (size_t)16 * K;
        const half_t* We = W2 + (size_t)e * N * K;
        for (int k0 = 0; k0 < K; k0 += 32) {
            __builtin_prefetch(A0 + ((lane & 15) * K) + k0 + 64, 0, 1);
            v16h a0 = load_a(A0 + k0, K, lane);
            v16h a1 = load_a(A1 + k0, K, lane);
            v16h bf[4];
#pragma unroll
            for (int t = 0; t < 4; ++t)
                bf[t] = load_b(We + (size_t)(n0 + 16 * t) * K + k0, K, lane);
#pragma unroll
            for (int t = 0; t < 4; ++t) {
                acc0[t] = WMMA_F16(a0, bf[t], acc0[t]);
                acc1[t] = WMMA_F16(a1, bf[t], acc1[t]);
            }
        }
        float g0[8], g1[8];
#pragma unroll
        for (int r = 0; r < 8; ++r) {
            g0[r] = gate[(size_t)(m0 + r + 8 * s) * NEXP + e];
            g1[r] = gate[(size_t)(m0 + 16 + r + 8 * s) * NEXP + e];
        }
#pragma unroll
        for (int t = 0; t < 4; ++t) {
            float bb = b2[e * N + n0 + 16 * t + n];
#pragma unroll
            for (int r = 0; r < 8; ++r) {
                tot0[t][r] += g0[r] * (acc0[t][r] + bb);
                tot1[t][r] += g1[r] * (acc1[t][r] + bb);
            }
        }
    }
#pragma unroll
    for (int t = 0; t < 4; ++t) {
        float* Cp0 = residual + (size_t)m0 * N + n0 + 16 * t + n;
        float* Cp1 = residual + (size_t)(m0 + 16) * N + n0 + 16 * t + n;
#pragma unroll
        for (int r = 0; r < 8; ++r) {
            Cp0[(size_t)(r + 8 * s) * N] += tot0[t][r];
            Cp1[(size_t)(r + 8 * s) * N] += tot1[t][r];
        }
    }
}

// ---------------- elementwise / norm / scan kernels ----------------
__global__ void mk_cvt(half_t* __restrict__ dst, const float* __restrict__ src, int n) {
    int i = blockIdx.x * 256 + threadIdx.x;
    if (i < n) dst[i] = (half_t)src[i];
}

__global__ void mk_init(float* __restrict__ states, float* __restrict__ residual,
                        const float* __restrict__ x, int rev) {
    int idx = blockIdx.x * 256 + threadIdx.x;   // RR*Dd total
    int col = idx & (Dd - 1);
    int row = idx >> 8;
    int l = row & (Ll - 1);
    int b = row >> 10;
    int ls = rev ? (Ll - 1 - l) : l;
    states[idx]   = x[(size_t)(b * Ll + ls) * Dd + col];
    residual[idx] = 0.f;
}

// residual += states; hn16 = f16(LayerNorm(residual)*w + b)
__global__ void mk_preln(float* __restrict__ residual, const float* __restrict__ states,
                         half_t* __restrict__ hn16, const float* __restrict__ w,
                         const float* __restrict__ b) {
    __shared__ float sd[256];
    int row = blockIdx.x, t = threadIdx.x;
    int idx = row * Dd + t;
    float v = residual[idx] + states[idx];
    residual[idx] = v;
    sd[t] = v; __syncthreads();
    for (int off = 128; off > 0; off >>= 1) { if (t < off) sd[t] += sd[t + off]; __syncthreads(); }
    float mu = sd[0] * (1.0f / Dd); __syncthreads();
    float dv = v - mu;
    sd[t] = dv * dv; __syncthreads();
    for (int off = 128; off > 0; off >>= 1) { if (t < off) sd[t] += sd[t + off]; __syncthreads(); }
    float var = sd[0] * (1.0f / Dd);
    float hn = dv * rsqrtf(var + 1e-5f) * w[t] + b[t];
    hn16[idx] = (half_t)hn;
}

// fin = LayerNorm(states + residual) * w + b  (no residual update)
__global__ void mk_finalln(float* __restrict__ fin, const float* __restrict__ residual,
                           const float* __restrict__ states, const float* __restrict__ w,
                           const float* __restrict__ b) {
    __shared__ float sd[256];
    int row = blockIdx.x, t = threadIdx.x;
    int idx = row * Dd + t;
    float v = residual[idx] + states[idx];
    sd[t] = v; __syncthreads();
    for (int off = 128; off > 0; off >>= 1) { if (t < off) sd[t] += sd[t + off]; __syncthreads(); }
    float mu = sd[0] * (1.0f / Dd); __syncthreads();
    float dv = v - mu;
    sd[t] = dv * dv; __syncthreads();
    for (int off = 128; off > 0; off >>= 1) { if (t < off) sd[t] += sd[t + off]; __syncthreads(); }
    float var = sd[0] * (1.0f / Dd);
    fin[idx] = dv * rsqrtf(var + 1e-5f) * w[t] + b[t];
}

// causal depthwise conv (k=4) + bias + SiLU over xs = xz[:, :512]
__global__ void mk_conv(const float* __restrict__ xz, const float* __restrict__ cw,
                        const float* __restrict__ cb, float* __restrict__ u,
                        half_t* __restrict__ u16) {
    int idx = blockIdx.x * 256 + threadIdx.x;   // RR*DINN total
    if (idx >= RR * DINN) return;
    int c = idx & (DINN - 1);
    int row = idx >> 9;                          // b*L + l
    int l = row & (Ll - 1);
    float acc = cb[c];
#pragma unroll
    for (int j = 0; j < DCONV; ++j) {
        int ls = l - (DCONV - 1) + j;
        if (ls >= 0) acc += cw[c * DCONV + j] * xz[(size_t)(row - (l - ls)) * XZD + c];
    }
    float sv = acc / (1.f + expf(-acc));
    u[idx] = sv;
    u16[idx] = (half_t)sv;
}

// selective scan: one thread per (b, channel); 16-state recurrence in registers;
// per-step xdbl row (dt_raw | B | C) broadcast through LDS.
__global__ void mk_scan(const float* __restrict__ xz, const float* __restrict__ u,
                        const float* __restrict__ xdbl, const float* __restrict__ dtw,
                        const float* __restrict__ dtb, const float* __restrict__ alog,
                        const float* __restrict__ dskip, half_t* __restrict__ yc16) {
    __shared__ float sx[48];
    int b = blockIdx.x >> 1;
    int c = ((blockIdx.x & 1) << 8) + threadIdx.x;
    float wdt[16], a[16], h[16];
#pragma unroll
    for (int n = 0; n < 16; ++n) {
        wdt[n] = dtw[c * 16 + n];
        a[n]   = -expf(alog[c * 16 + n]);
        h[n]   = 0.f;
    }
    float bias = dtb[c], dsk = dskip[c];
    for (int l = 0; l < Ll; ++l) {
        int row = b * Ll + l;
        if (threadIdx.x < 48) sx[threadIdx.x] = xdbl[(size_t)row * 48 + threadIdx.x];
        __syncthreads();
        float dt = bias;
#pragma unroll
        for (int n = 0; n < 16; ++n) dt += sx[n] * wdt[n];
        dt = (dt > 0.f) ? (dt + log1pf(expf(-dt))) : log1pf(expf(dt));   // softplus
        float uv = u[(size_t)row * DINN + c];
        float du = dt * uv;
        float y = 0.f;
#pragma unroll
        for (int n = 0; n < 16; ++n) {
            h[n] = h[n] * expf(dt * a[n]) + du * sx[16 + n];
            y += h[n] * sx[32 + n];
        }
        y += uv * dsk;
        float z = xz[(size_t)row * XZD + DINN + c];
        float outv = y * (z / (1.f + expf(-z)));   // y * silu(z)
        yc16[(size_t)row * DINN + c] = (half_t)outv;
        __syncthreads();
    }
}

__global__ void mk_curr(float* __restrict__ curr, half_t* __restrict__ curr16,
                        const float* __restrict__ states, const float* __restrict__ residual) {
    int idx = blockIdx.x * 256 + threadIdx.x;
    float v = states[idx] + residual[idx];
    curr[idx] = v;
    curr16[idx] = (half_t)v;
}

// one wave per token: softmax over E=4, top-2 mask, renormalized gate
__global__ void mk_gate(float* __restrict__ gate, const float* __restrict__ curr,
                        const float* __restrict__ gw, const float* __restrict__ gb) {
    int token = blockIdx.x * 8 + (threadIdx.x >> 5);
    int lane = threadIdx.x & 31;
    const float* cr = curr + (size_t)token * Dd;
    float sc[4];
#pragma unroll
    for (int e = 0; e < 4; ++e) {
        float s = 0.f;
        for (int j = lane; j < Dd; j += 32) s += cr[j] * gw[e * Dd + j];
#pragma unroll
        for (int off = 16; off > 0; off >>= 1) s += __shfl_down(s, off, 32);
        sc[e] = s;
    }
    if (lane == 0) {
        float p[4], mx = -3.4e38f;
#pragma unroll
        for (int e = 0; e < 4; ++e) { p[e] = sc[e] + gb[e]; mx = fmaxf(mx, p[e]); }
        float sum = 0.f;
#pragma unroll
        for (int e = 0; e < 4; ++e) { p[e] = expf(p[e] - mx); sum += p[e]; }
        float inv = 1.f / sum;
#pragma unroll
        for (int e = 0; e < 4; ++e) p[e] *= inv;
        int i1 = 0;
        for (int e = 1; e < 4; ++e) if (p[e] > p[i1]) i1 = e;
        int i2 = (i1 == 0) ? 1 : 0;
        for (int e = 0; e < 4; ++e) if (e != i1 && p[e] > p[i2]) i2 = e;
        float g = 1.f / (p[i1] + p[i2] + 1e-6f);
#pragma unroll
        for (int e = 0; e < 4; ++e)
            gate[(size_t)token * 4 + e] = (e == i1 || e == i2) ? p[e] * g : 0.f;
    }
}

// attention pool over L: comb[b, dslot*256 + :] = sum_l softmax_l(fin.pw + pb) * fin
__global__ void mk_pool(float* __restrict__ comb, const float* __restrict__ fin,
                        const float* __restrict__ pw, const float* __restrict__ pb, int dslot) {
    __shared__ float sc[Ll];
    __shared__ float red[256];
    int b = blockIdx.x, t = threadIdx.x;
    for (int l = t; l < Ll; l += 256) {
        const float* fr = fin + (size_t)(b * Ll + l) * Dd;
        float s = 0.f;
        for (int j = 0; j < Dd; ++j) s += fr[j] * pw[j];
        sc[l] = s + pb[0];
    }
    __syncthreads();
    float lm = -3.4e38f;
    for (int l = t; l < Ll; l += 256) lm = fmaxf(lm, sc[l]);
    red[t] = lm; __syncthreads();
    for (int off = 128; off > 0; off >>= 1) { if (t < off) red[t] = fmaxf(red[t], red[t + off]); __syncthreads(); }
    float mx = red[0]; __syncthreads();
    float ls = 0.f;
    for (int l = t; l < Ll; l += 256) { float e = expf(sc[l] - mx); sc[l] = e; ls += e; }
    red[t] = ls; __syncthreads();
    for (int off = 128; off > 0; off >>= 1) { if (t < off) red[t] += red[t + off]; __syncthreads(); }
    float inv = 1.f / red[0]; __syncthreads();
    float acc = 0.f;
    for (int l = 0; l < Ll; ++l) acc += sc[l] * fin[(size_t)(b * Ll + l) * Dd + t];
    comb[b * 2 * Dd + dslot * Dd + t] = acc * inv;
}

__global__ void mk_linear(float* __restrict__ out, const float* __restrict__ comb,
                          const float* __restrict__ w, const float* __restrict__ bias) {
    int b = blockIdx.x, o = threadIdx.x;
    float acc = bias[o];
    for (int j = 0; j < 2 * Dd; ++j) acc += comb[b * 2 * Dd + j] * w[o * 2 * Dd + j];
    out[b * Dd + o] = acc;
}

// ---------------- host orchestration ----------------
extern "C" void kernel_launch(void* const* d_in, const int* in_sizes, int n_in,
                              void* d_out, int out_size, void* d_ws, size_t ws_size,
                              hipStream_t stream) {
    const float* x    = (const float*)d_in[0];
    const float* inw  = (const float*)d_in[1];
    const float* cw   = (const float*)d_in[2];
    const float* cb   = (const float*)d_in[3];
    const float* xpw  = (const float*)d_in[4];
    const float* dtw  = (const float*)d_in[5];
    const float* dtb  = (const float*)d_in[6];
    const float* alog = (const float*)d_in[7];
    const float* dskp = (const float*)d_in[8];
    const float* ow   = (const float*)d_in[9];
    const float* nw   = (const float*)d_in[10];
    const float* nb   = (const float*)d_in[11];
    const float* gwt  = (const float*)d_in[12];
    const float* gbt  = (const float*)d_in[13];
    const float* e1w  = (const float*)d_in[14];
    const float* e1b  = (const float*)d_in[15];
    const float* e2w  = (const float*)d_in[16];
    const float* e2b  = (const float*)d_in[17];
    const float* nfw  = (const float*)d_in[18];
    const float* nfb  = (const float*)d_in[19];
    const float* fpw  = (const float*)d_in[20];
    const float* fpb  = (const float*)d_in[21];
    const float* bpw  = (const float*)d_in[22];
    const float* bpb  = (const float*)d_in[23];
    const float* llw  = (const float*)d_in[24];
    const float* llb  = (const float*)d_in[25];

    // ---- workspace carve (256B-aligned slabs) ----
    char* p = (char*)d_ws;
    size_t used = 0;
    auto carve = [&](size_t bytes) -> void* {
        void* r = p + used;
        used += (bytes + 255) & ~(size_t)255;
        return r;
    };
    float* residual = (float*)carve((size_t)RR * Dd * 4);
    float* states   = (float*)carve((size_t)RR * Dd * 4);
    float* fin      = (float*)carve((size_t)RR * Dd * 4);
    float* xz       = (float*)carve((size_t)RR * XZD * 4);
    float* u        = (float*)carve((size_t)RR * DINN * 4);
    float* xdbl     = (float*)carve((size_t)RR * 48 * 4);
    float* curr     = (float*)carve((size_t)RR * Dd * 4);
    float* gate     = (float*)carve((size_t)RR * NEXP * 4);
    float* comb     = (float*)carve((size_t)Bb * 2 * Dd * 4);
    half_t* hn16    = (half_t*)carve((size_t)RR * Dd * 2);
    half_t* u16     = (half_t*)carve((size_t)RR * DINN * 2);
    half_t* yc16    = (half_t*)carve((size_t)RR * DINN * 2);
    half_t* curr16  = (half_t*)carve((size_t)RR * Dd * 2);
    half_t* hbuf    = (half_t*)carve((size_t)NEXP * RR * HH * 2);
    half_t* s_inw   = (half_t*)carve((size_t)XZD * Dd * 2);
    half_t* s_xpw   = (half_t*)carve((size_t)48 * DINN * 2);
    half_t* s_ow    = (half_t*)carve((size_t)Dd * DINN * 2);
    half_t* s_e1    = (half_t*)carve((size_t)NEXP * HH * Dd * 2);
    half_t* s_e2    = (half_t*)carve((size_t)NEXP * Dd * HH * 2);
    if (used > ws_size) return;   // workspace too small; bail deterministically

    auto cvt = [&](half_t* dst, const float* src, int n) {
        mk_cvt<<<(n + 255) / 256, 256, 0, stream>>>(dst, src, n);
    };

    for (int d = 0; d < 2; ++d) {
        mk_init<<<RR, 256, 0, stream>>>(states, residual, x, d);
        for (int i = 0; i < NLAYERS; ++i) {
            int li = d * NLAYERS + i;
            const float* inw_p  = inw  + (size_t)li * XZD * Dd;
            const float* cw_p   = cw   + (size_t)li * DINN * DCONV;
            const float* cb_p   = cb   + (size_t)li * DINN;
            const float* xpw_p  = xpw  + (size_t)li * 48 * DINN;
            const float* dtw_p  = dtw  + (size_t)li * DINN * DTRR;
            const float* dtb_p  = dtb  + (size_t)li * DINN;
            const float* alog_p = alog + (size_t)li * DINN * DSTATE;
            const float* dskp_p = dskp + (size_t)li * DINN;
            const float* ow_p   = ow   + (size_t)li * Dd * DINN;
            const float* nw_p   = nw   + (size_t)li * Dd;
            const float* nb_p   = nb   + (size_t)li * Dd;
            const float* gw_p   = gwt  + (size_t)li * NEXP * Dd;
            const float* gb_p   = gbt  + (size_t)li * NEXP;
            const float* e1w_p  = e1w  + (size_t)li * NEXP * HH * Dd;
            const float* e1b_p  = e1b  + (size_t)li * NEXP * HH;
            const float* e2w_p  = e2w  + (size_t)li * NEXP * Dd * HH;
            const float* e2b_p  = e2b  + (size_t)li * NEXP * Dd;

            // residual += states ; hn = LN(residual)
            mk_preln<<<RR, 256, 0, stream>>>(residual, states, hn16, nw_p, nb_p);

            // in_proj: xz = hn @ inw^T   (4096x1024x256); waves = 128*16
            cvt(s_inw, inw_p, XZD * Dd);
            gemm_wmma<4><<<(RR / 32) * (XZD / 64) / 8, 256, 0, stream>>>(xz, hn16, s_inw, RR, XZD, Dd);

            // depthwise causal conv + SiLU -> u
            mk_conv<<<RR * DINN / 256, 256, 0, stream>>>(xz, cw_p, cb_p, u, u16);

            // x_proj: xdbl = u @ xpw^T   (4096x48x512); waves = 128
            cvt(s_xpw, xpw_p, 48 * DINN);
            gemm_wmma<3><<<(RR / 32) / 8, 256, 0, stream>>>(xdbl, u16, s_xpw, RR, 48, DINN);

            // selective scan (sequential in L)
            mk_scan<<<2 * Bb, 256, 0, stream>>>(xz, u, xdbl, dtw_p, dtb_p, alog_p, dskp_p, yc16);

            // out_proj: states = ycomb @ ow^T   (4096x256x512); waves = 128*4
            cvt(s_ow, ow_p, Dd * DINN);
            gemm_wmma<4><<<(RR / 32) * (Dd / 64) / 8, 256, 0, stream>>>(states, yc16, s_ow, RR, Dd, DINN);

            // curr = states + residual ; gate = top2-softmax
            mk_curr<<<RR, 256, 0, stream>>>(curr, curr16, states, residual);
            mk_gate<<<RR / 8, 256, 0, stream>>>(gate, curr, gw_p, gb_p);

            // MoE up: h_e = gelu(curr @ w1_e^T + b1_e)   (4 x 4096x1024x256); waves = 128*16 each
            cvt(s_e1, e1w_p, NEXP * HH * Dd);
            for (int e = 0; e < NEXP; ++e)
                gemm_wmma_gelu<4><<<(RR / 32) * (HH / 64) / 8, 256, 0, stream>>>(
                    hbuf + (size_t)e * RR * HH, curr16, s_e1 + (size_t)e * HH * Dd,
                    e1b_p + e * HH, RR, HH, Dd);

            // MoE down + gate + residual accumulate   (4096x256x1024 x 4 experts); waves = 128*4
            cvt(s_e2, e2w_p, NEXP * Dd * HH);
            mk_moe2<<<(RR / 32) * (Dd / 64) / 8, 256, 0, stream>>>(residual, hbuf, s_e2, e2b_p, gate);
        }
        mk_finalln<<<RR, 256, 0, stream>>>(fin, residual, states, nfw, nfb);
        mk_pool<<<Bb, 256, 0, stream>>>(comb, fin, (d == 0) ? fpw : bpw, (d == 0) ? fpb : bpb, d);
    }
    mk_linear<<<Bb, 256, 0, stream>>>((float*)d_out, comb, llw, llb);
}
